// ActorHead_44994077392970
// MI455X (gfx1250) — compile-verified, matrix operand order checked by reading, not run on hardware
//
#include <hip/hip_runtime.h>

// ---------------------------------------------------------------------------
// Problem constants (reference: N=4096, M=4096, D=1024, H=8 -> dk=128)
// ---------------------------------------------------------------------------
#define N_DIM 4096
#define M_DIM 4096
#define D_DIM 1024
#define H_DIM 8
#define DK    128
#define BC    32          // attention column-tile (K/V rows per iteration)

typedef __bf16 bf16;
typedef __attribute__((ext_vector_type(16))) __bf16 v16bf;
typedef __attribute__((ext_vector_type(8)))  __bf16 v8bf;
typedef __attribute__((ext_vector_type(8)))  float  v8f;

typedef __attribute__((ext_vector_type(4))) unsigned int u32x4;
typedef __attribute__((ext_vector_type(4))) int          i32x4;
typedef __attribute__((ext_vector_type(8))) int          i32x8;

#if __has_builtin(__builtin_amdgcn_tensor_load_to_lds) && \
    __has_builtin(__builtin_amdgcn_s_wait_tensorcnt)
#define USE_TDM 1
#else
#define USE_TDM 0
#endif

// Assemble a 16-element bf16 WMMA fragment from two contiguous 8-element
// (16-byte) chunks.  Per the CDNA5 16-bit A/B layout, a lane holds
// K = {koff..koff+7} in VGPR0-3 and K = {16+koff..16+koff+7} in VGPR4-7
// (koff = 0 for lanes 0-15, 8 for lanes 16-31), and v16bf element order maps
// exactly onto that VGPR packing.
union FragU {
  v16bf v;
  struct { v8bf lo; v8bf hi; } h;
};
__device__ __forceinline__ v16bf make_frag(const bf16* p0, const bf16* p1) {
  FragU u;
  u.h.lo = *(const v8bf*)p0;
  u.h.hi = *(const v8bf*)p1;
  return u.v;
}

// ---------------------------------------------------------------------------
// Kernel 1: projection GEMM   Out[r, j] = sum_d X[r, d] * W[j, d]   (bf16 out)
// 128x128 output tile per workgroup, 8 waves (wave32), bf16 WMMA 16x16x32.
// B-matrix column j == row j of W, so A and B staging are symmetric.
// ---------------------------------------------------------------------------
__global__ __launch_bounds__(256)
void proj_kernel(const float* __restrict__ X, const float* __restrict__ W,
                 bf16* __restrict__ Out)
{
  __shared__ bf16 As[128][32];   // 8 KB
  __shared__ bf16 Bs[128][32];   // 8 KB

  const int tid  = threadIdx.x;
  const int lane = tid & 31;
  const int wave = tid >> 5;
  const int row0 = blockIdx.x * 128;
  const int col0 = blockIdx.y * 128;
  const int wr   = (wave >> 1) * 32;  // wave's row offset inside tile
  const int wc   = (wave & 1) * 64;   // wave's col offset inside tile
  const int hi   = lane >> 4;
  const int l15  = lane & 15;
  const int koff = hi * 8;

  v8f acc[2][4] = {};

  for (int k0 = 0; k0 < D_DIM; k0 += 32) {
    // Stage 128x32 f32 -> bf16 for A and B (1024 float4 each, 4 per thread).
    for (int i = 0; i < 4; ++i) {
      int idx = tid + i * 256;           // 0..1023
      int r   = idx >> 3;                // 8 float4 per 32-col row
      int c   = (idx & 7) * 4;
      float4 a = *(const float4*)&X[(size_t)(row0 + r) * D_DIM + k0 + c];
      As[r][c + 0] = (bf16)a.x; As[r][c + 1] = (bf16)a.y;
      As[r][c + 2] = (bf16)a.z; As[r][c + 3] = (bf16)a.w;
      float4 b = *(const float4*)&W[(size_t)(col0 + r) * D_DIM + k0 + c];
      Bs[r][c + 0] = (bf16)b.x; Bs[r][c + 1] = (bf16)b.y;
      Bs[r][c + 2] = (bf16)b.z; Bs[r][c + 3] = (bf16)b.w;
    }
    __syncthreads();

    v16bf afrag[2];
    for (int i = 0; i < 2; ++i) {
      int r = wr + i * 16 + l15;
      afrag[i] = make_frag(&As[r][koff], &As[r][16 + koff]);
    }
    for (int j = 0; j < 4; ++j) {
      int cc = wc + j * 16 + l15;
      v16bf bfrag = make_frag(&Bs[cc][koff], &Bs[cc][16 + koff]);
      for (int i = 0; i < 2; ++i)
        acc[i][j] = __builtin_amdgcn_wmma_f32_16x16x32_bf16(
            false, afrag[i], false, bfrag, (short)0, acc[i][j], false, false);
    }
    __syncthreads();
  }

  // C-layout store: lane 0-15 -> N=lane, VGPR r -> M=r (+8 for upper half).
  for (int i = 0; i < 2; ++i)
    for (int j = 0; j < 4; ++j)
      for (int r = 0; r < 8; ++r) {
        int orow = row0 + wr + i * 16 + r + hi * 8;
        int ocol = col0 + wc + j * 16 + l15;
        Out[(size_t)orow * D_DIM + ocol] = (bf16)acc[i][j][r];
      }
}

// ---------------------------------------------------------------------------
// Kernel 2: flash attention per (head, 64 q-rows).  4 waves x 16 rows.
// K tiles are fetched by the Tensor Data Mover (tensor_load_to_lds +
// s_wait_tensorcnt); V tiles are transposed into LDS by the VALU path;
// online softmax lives in VGPRs; all GEMM stages are bf16 WMMA.
// ---------------------------------------------------------------------------
__global__ __launch_bounds__(128)
void attn_kernel(const bf16* __restrict__ Qb, const bf16* __restrict__ Kb,
                 const bf16* __restrict__ Vb, const int* __restrict__ mask,
                 const float* __restrict__ weight, float* __restrict__ ctx)
{
  __shared__ bf16 Ks[BC][DK];        // 8 KB  (row-major K tile, TDM target)
  __shared__ bf16 Vt[DK][BC + 8];    // 10 KB (V transposed; stride 80 B, 16B-aligned)
  __shared__ bf16 Ps[4][16][BC];     // 4 KB  (per-wave probability tile)

  const int tid   = threadIdx.x;
  const int lane  = tid & 31;
  const int wave  = tid >> 5;
  const int hi    = lane >> 4;
  const int l15   = lane & 15;
  const int koff  = hi * 8;
  const int head  = blockIdx.y;
  const int hc    = head * DK;                    // head's column base in D
  const int qrow0 = blockIdx.x * 64 + wave * 16;

  // Persistent Q fragments: 16 rows x dk=128 as 4 K-chunks of 32.
  v16bf aq[4];
  {
    const bf16* qrow = Qb + (size_t)(qrow0 + l15) * D_DIM + hc;
    for (int kc = 0; kc < 4; ++kc)
      aq[kc] = make_frag(qrow + kc * 32 + koff, qrow + kc * 32 + 16 + koff);
  }

  float mrow[8], lrow[8];
  v8f   o[8] = {};
  for (int r = 0; r < 8; ++r) { mrow[r] = -3.0e38f; lrow[r] = 0.0f; }
  const float scale = 0.08838834764831845f;       // 1/sqrt(dk)

#if USE_TDM
  const unsigned ldsK = (unsigned)(size_t)(&Ks[0][0]);   // wave-relative LDS byte addr
#endif

  for (int jb = 0; jb < M_DIM / BC; ++jb) {
    const int j0 = jb * BC;

    // Prefetch next V tile while we work on this one (global_prefetch_b8).
    if (jb + 1 < M_DIM / BC) {
      int pr = tid >> 2, pc = (tid & 3) * 32;
      __builtin_prefetch(&Vb[(size_t)(j0 + BC + pr) * D_DIM + hc + pc], 0, 1);
    }

#if USE_TDM
    // --- TDM: DMA the 32x128 bf16 K tile (row stride D_DIM elems) into Ks.
    // D# per CDNA5 ISA ch.8: group0 = {count|flags, lds_addr, gaddr_lo,
    // gaddr_hi|type=2}; group1 packs data_size=2B, tensor_dim0=128,
    // tensor_dim1=32, tile 128x32, tensor_dim0_stride=1024.
    if (wave == 0) {
      unsigned long long ga =
          (unsigned long long)(size_t)(Kb + (size_t)j0 * D_DIM + hc);
      u32x4 g0 = { 1u,                                  // count=1, user desc
                   ldsK,                                // D#.lds_addr
                   (unsigned)ga,                        // global_addr[31:0]
                   (unsigned)((ga >> 32) & 0x01FFFFFFu) | (2u << 30) }; // +type=2
      i32x8 g1 = { (int)0x00010000u,                    // data_size=1 (2 bytes)
                   (int)(128u << 16),                   // tensor_dim0 = 128 (lo16)
                   (int)(32u  << 16),                   // td0 hi16=0 | tensor_dim1=32
                   (int)(128u << 16),                   // td1 hi16=0 | tile_dim0=128
                   32,                                  // tile_dim1=32 | tile_dim2=0
                   1024,                                // tensor_dim0_stride lo32
                   0, 0 };                              // stride hi | dim1_stride
      i32x4 g2 = { 0, 0, 0, 0 };
      i32x4 g3 = { 0, 0, 0, 0 };
#if defined(__clang_major__) && (__clang_major__ >= 23)
      i32x8 gz = { 0, 0, 0, 0, 0, 0, 0, 0 };
      __builtin_amdgcn_tensor_load_to_lds(g0, g1, g2, g3, gz, 0);
#else
      __builtin_amdgcn_tensor_load_to_lds(g0, g1, g2, g3, 0);
#endif
      __builtin_amdgcn_s_wait_tensorcnt(0);
    }
    // V tile: transpose-stage via VALU (TDM cannot transpose).
    for (int i = 0; i < 4; ++i) {
      int idx = tid + i * 128;
      int r   = idx >> 4;
      int c   = (idx & 15) * 8;
      v8bf vv = *(const v8bf*)&Vb[(size_t)(j0 + r) * D_DIM + hc + c];
      for (int e = 0; e < 8; ++e) Vt[c + e][r] = vv[e];
    }
#else
    // Fallback: stage K (row-major) and V (transposed) tiles manually.
    for (int i = 0; i < 4; ++i) {
      int idx = tid + i * 128;            // 0..511 chunks of 8 bf16
      int r   = idx >> 4;                 // 0..31
      int c   = (idx & 15) * 8;           // 0..120
      v8bf kv = *(const v8bf*)&Kb[(size_t)(j0 + r) * D_DIM + hc + c];
      *(v8bf*)&Ks[r][c] = kv;
      v8bf vv = *(const v8bf*)&Vb[(size_t)(j0 + r) * D_DIM + hc + c];
      for (int e = 0; e < 8; ++e) Vt[c + e][r] = vv[e];
    }
#endif
    __syncthreads();

    // S = (Q Kt)/sqrt(dk): two 16x16 tiles covering 32 columns.
    v8f s[2];
    for (int ct = 0; ct < 2; ++ct) {
      v8f sv = {};
      int mr = ct * 16 + l15;             // B column m == K-tile row m
      for (int kc = 0; kc < 4; ++kc) {
        v16bf bk = make_frag(&Ks[mr][kc * 32 + koff],
                             &Ks[mr][kc * 32 + 16 + koff]);
        sv = __builtin_amdgcn_wmma_f32_16x16x32_bf16(
            false, aq[kc], false, bk, (short)0, sv, false, false);
      }
      s[ct] = sv;
    }

    // mask / weight in C-layout (lane = column, VGPR index = row).
    for (int ct = 0; ct < 2; ++ct) {
      int mcol = j0 + ct * 16 + l15;
      for (int r = 0; r < 8; ++r) {
        size_t off = (size_t)(qrow0 + r + hi * 8) * M_DIM + mcol;
        float sv = s[ct][r] * scale;
        sv = mask[off] ? sv : -1.0e9f;
        s[ct][r] = sv + weight[off];
      }
    }

    // Online softmax.  Rows live in VGPR lanes consistently (same half),
    // so row max/sum are 16-lane butterflies; rescale is lane-local.
    float pr0[8], pr1[8];
    for (int r = 0; r < 8; ++r) {
      float mx = fmaxf(s[0][r], s[1][r]);
      mx = fmaxf(mx, __shfl_xor(mx, 8));
      mx = fmaxf(mx, __shfl_xor(mx, 4));
      mx = fmaxf(mx, __shfl_xor(mx, 2));
      mx = fmaxf(mx, __shfl_xor(mx, 1));
      float mn    = fmaxf(mrow[r], mx);
      float alpha = __expf(mrow[r] - mn);
      float p0 = __expf(s[0][r] - mn);
      float p1 = __expf(s[1][r] - mn);
      float rs = p0 + p1;
      rs += __shfl_xor(rs, 8);
      rs += __shfl_xor(rs, 4);
      rs += __shfl_xor(rs, 2);
      rs += __shfl_xor(rs, 1);
      lrow[r] = lrow[r] * alpha + rs;
      mrow[r] = mn;
      for (int t = 0; t < 8; ++t) o[t][r] *= alpha;
      pr0[r] = p0; pr1[r] = p1;
    }

    // C-layout -> A-layout conversion for P via per-wave LDS tile.
    for (int r = 0; r < 8; ++r) {
      Ps[wave][r + hi * 8][l15]      = (bf16)pr0[r];
      Ps[wave][r + hi * 8][16 + l15] = (bf16)pr1[r];
    }
    // Same-wave LDS ordering makes the reload safe without a barrier.
    v16bf ap = make_frag(&Ps[wave][l15][koff], &Ps[wave][l15][16 + koff]);

    // O += P @ V : B column d == row d of transposed V tile (contiguous).
    for (int nt = 0; nt < 8; ++nt) {
      int dr = nt * 16 + l15;
      v16bf bv = make_frag(&Vt[dr][koff], &Vt[dr][16 + koff]);
      o[nt] = __builtin_amdgcn_wmma_f32_16x16x32_bf16(
          false, ap, false, bv, (short)0, o[nt], false, false);
    }
    __syncthreads();   // also protects Ks against next iteration's TDM write
  }

  // Normalize and write per-head context [h][N][dk].
  for (int nt = 0; nt < 8; ++nt)
    for (int r = 0; r < 8; ++r) {
      int nrow = qrow0 + r + hi * 8;
      int d    = nt * 16 + l15;
      ctx[((size_t)head * N_DIM + nrow) * DK + d] = o[nt][r] / lrow[r];
    }
}

// ---------------------------------------------------------------------------
// Kernel 3: topic_align = mean over heads; influence = 1.0 exactly
// (softmax rows sum to 1, so mean_h sum_m p == 1).
// ---------------------------------------------------------------------------
__global__ __launch_bounds__(256)
void reduce_kernel(const float* __restrict__ ctx, float* __restrict__ out)
{
  const int total = N_DIM * DK;
  int gid = blockIdx.x * 256 + threadIdx.x;
  if (gid < total) {
    float sum = 0.0f;
    for (int h = 0; h < H_DIM; ++h) sum += ctx[(size_t)h * total + gid];
    out[gid] = sum * (1.0f / H_DIM);
  } else if (gid < total + N_DIM) {
    out[gid] = 1.0f;
  }
}

// ---------------------------------------------------------------------------
// Launch
// ---------------------------------------------------------------------------
extern "C" void kernel_launch(void* const* d_in, const int* in_sizes, int n_in,
                              void* d_out, int out_size, void* d_ws, size_t ws_size,
                              hipStream_t stream)
{
  (void)in_sizes; (void)n_in; (void)out_size; (void)ws_size;
  const float* a_z    = (const float*)d_in[0];
  const float* bv_z   = (const float*)d_in[1];
  const int*   mask   = (const int*)d_in[2];
  const float* weight = (const float*)d_in[3];
  const float* Wq     = (const float*)d_in[4];
  const float* Wk     = (const float*)d_in[5];
  const float* Wv     = (const float*)d_in[6];
  float* out = (float*)d_out;

  // Workspace: Q/K/V bf16 (8 MB each) + per-head context f32 (16 MB) = 40 MB.
  char* ws = (char*)d_ws;
  bf16*  Qb  = (bf16*)(ws);
  bf16*  Kb  = (bf16*)(ws + ((size_t)8  << 20));
  bf16*  Vb  = (bf16*)(ws + ((size_t)16 << 20));
  float* ctx = (float*)(ws + ((size_t)24 << 20));

  dim3 pgrid(N_DIM / 128, D_DIM / 128);
  proj_kernel<<<pgrid, 256, 0, stream>>>(a_z,  Wq, Qb);
  proj_kernel<<<pgrid, 256, 0, stream>>>(bv_z, Wk, Kb);
  proj_kernel<<<pgrid, 256, 0, stream>>>(bv_z, Wv, Vb);

  dim3 agrid(N_DIM / 64, H_DIM);
  attn_kernel<<<agrid, 128, 0, stream>>>(Qb, Kb, Vb, mask, weight, ctx);

  int total = N_DIM * DK + N_DIM;
  reduce_kernel<<<(total + 255) / 256, 256, 0, stream>>>(ctx, out);
}
